// CrossAttention_7928509629236
// MI455X (gfx1250) — matrix operand order repaired and checked by reading, not compiled, not run on hardware
//
#include <hip/hip_runtime.h>
#include <hip/hip_bf16.h>

// ---------------- problem constants ----------------
static constexpr int kB  = 8;
static constexpr int kSQ = 1024;
static constexpr int kSK = 1024;
static constexpr int kQD = 2048;
static constexpr int kKVD = 1024;
static constexpr int kE  = 2048;
static constexpr int kH  = 16;
static constexpr int kHD = 128;

typedef __attribute__((ext_vector_type(16))) __bf16 v16bf;
typedef __attribute__((ext_vector_type(8)))  __bf16 v8bf;
typedef __attribute__((ext_vector_type(4)))  __bf16 v4bf;
typedef __attribute__((ext_vector_type(8)))  float  v8f;
typedef __attribute__((ext_vector_type(4)))  float  v4f;
typedef __attribute__((ext_vector_type(4)))  int    v4i;

union AFrag { v16bf v; v8bf h[2]; __bf16 e[16]; };

static __device__ __forceinline__ v8f vzero8() {
    v8f z = {0.f,0.f,0.f,0.f,0.f,0.f,0.f,0.f};
    return z;
}

// ---------------- CDNA5 async global->LDS copy (ASYNCcnt path) ----------------
// 16 bytes per lane, data lands directly in LDS without touching VGPRs.
// Builtin signature (from hipcc diagnostic): param1 = v4i in global (AS1),
// param2 = v4i in LDS (AS3), then two immediate ints (offset, cpol).
static __device__ __forceinline__ void async_copy16(void* lds_dst, const void* gsrc) {
#if __has_builtin(__builtin_amdgcn_global_load_async_to_lds_b128)
    __builtin_amdgcn_global_load_async_to_lds_b128(
        (__attribute__((address_space(1))) v4i*)((void*)gsrc),
        (__attribute__((address_space(3))) v4i*)lds_dst, 0, 0);
#else
    const unsigned loff =
        (unsigned)(__SIZE_TYPE__)((__attribute__((address_space(3))) void*)lds_dst);
    asm volatile("global_load_async_to_lds_b128 %0, %1, off"
                 :: "v"(loff), "v"(gsrc) : "memory");
#endif
}

template <int N>
static __device__ __forceinline__ void wait_async_le() {
#if __has_builtin(__builtin_amdgcn_s_wait_asynccnt)
    __builtin_amdgcn_s_wait_asynccnt((unsigned short)N);
#else
    asm volatile("s_wait_asynccnt %0" :: "i"(N) : "memory");
#endif
}

// ---------------- fp32 -> bf16 conversion ----------------
__global__ void cvt_f32_bf16_kernel(const float* __restrict__ src,
                                    __bf16* __restrict__ dst, int n4) {
    int i = blockIdx.x * blockDim.x + threadIdx.x;
    if (i < n4) {
        v4f x = ((const v4f*)src)[i];
        v4bf y;
        y[0] = (__bf16)x[0]; y[1] = (__bf16)x[1];
        y[2] = (__bf16)x[2]; y[3] = (__bf16)x[3];
        ((v4bf*)dst)[i] = y;
    }
}

// ---------------- GEMM: C[M,N] = A[M,K] * W[N,K]^T + bias[N] ----------------
// block = 256 threads = 8 wave32 waves; tile 256x128, K-step 32,
// double-buffered LDS filled by async global->LDS copies.
// wave w owns rows [w*32, w*32+32) x all 128 cols -> 2x8 WMMA tiles.
template <bool OUT_BF16>
__global__ __launch_bounds__(256)
void gemm_nt_kernel(const __bf16* __restrict__ A, const __bf16* __restrict__ W,
                    const float* __restrict__ bias, void* __restrict__ Cout,
                    int M, int N, int K) {
    __shared__ __bf16 As[2][256][40];   // 32 K-elems + 8 pad (row = 80B)
    __shared__ __bf16 Ws[2][128][40];

    const int t    = threadIdx.x;
    const int w    = t >> 5;
    const int lane = t & 31;
    const int ll   = lane & 15;
    const int lh   = lane >> 4;
    const int m0   = blockIdx.y * 256;
    const int n0   = blockIdx.x * 128;
    const int wm   = w * 32;
    const int lr   = t >> 1;
    const int lc   = (t & 1) * 16;

    v8f acc[2][8];
    for (int a = 0; a < 2; ++a)
        for (int j = 0; j < 8; ++j) acc[a][j] = vzero8();

    const __bf16* aRow0 = A + (size_t)(m0 + lr) * K + lc;
    const __bf16* aRow1 = A + (size_t)(m0 + 128 + lr) * K + lc;
    const __bf16* wRow  = W + (size_t)(n0 + lr) * K + lc;

    auto stage = [&](int buf, int k0) {   // 6 async b128 copies per thread
        async_copy16(&As[buf][lr][lc],           aRow0 + k0);
        async_copy16(&As[buf][lr][lc + 8],       aRow0 + k0 + 8);
        async_copy16(&As[buf][128 + lr][lc],     aRow1 + k0);
        async_copy16(&As[buf][128 + lr][lc + 8], aRow1 + k0 + 8);
        async_copy16(&Ws[buf][lr][lc],           wRow + k0);
        async_copy16(&Ws[buf][lr][lc + 8],       wRow + k0 + 8);
    };

    const int nsteps = K / 32;
    stage(0, 0);
    for (int s = 0; s < nsteps; ++s) {
        const int cur = s & 1;
        if (s + 1 < nsteps) {
            __syncthreads();                 // all waves done reading buf cur^1
            stage(cur ^ 1, (s + 1) * 32);    // prefetch next tile (async)
            wait_async_le<6>();              // in-order: current tile's 6 done
        } else {
            wait_async_le<0>();
        }
        __syncthreads();                     // everyone's copies for buf[cur] landed

        AFrag af[2];
        for (int a = 0; a < 2; ++a) {
            const int row = wm + a * 16 + ll;
            af[a].h[0] = *(const v8bf*)(&As[cur][row][lh * 8]);       // K 0-7 / 8-15
            af[a].h[1] = *(const v8bf*)(&As[cur][row][16 + lh * 8]);  // K 16-23 / 24-31
        }
        for (int j = 0; j < 8; ++j) {
            AFrag bfj;
            const int col = j * 16 + ll;
            bfj.h[0] = *(const v8bf*)(&Ws[cur][col][lh * 8]);         // K 0-7 / 8-15
            bfj.h[1] = *(const v8bf*)(&Ws[cur][col][16 + lh * 8]);
            acc[0][j] = __builtin_amdgcn_wmma_f32_16x16x32_bf16(
                false, af[0].v, false, bfj.v, (short)0, acc[0][j], false, false);
            acc[1][j] = __builtin_amdgcn_wmma_f32_16x16x32_bf16(
                false, af[1].v, false, bfj.v, (short)0, acc[1][j], false, false);
        }
    }

    for (int j = 0; j < 8; ++j) {
        const int n   = n0 + j * 16 + ll;
        const float bv = bias[n];
        for (int a = 0; a < 2; ++a) {
            for (int r = 0; r < 8; ++r) {
                const size_t row = (size_t)(m0 + wm + a * 16 + r + 8 * lh);
                const float v = acc[a][j][r] + bv;
                if constexpr (OUT_BF16) ((__bf16*)Cout)[row * N + n] = (__bf16)v;
                else                    ((float*)Cout)[row * N + n]  = v;
            }
        }
    }
}

// ---------------- flash attention ----------------
// grid: (SQ/128, H, B). 256 threads = 8 waves; wave w owns 16 q-rows.
// SK processed in 64-key tiles; K tile async-copied to LDS, V tile transposed.
__global__ __launch_bounds__(256)
void flash_attn_kernel(const __bf16* __restrict__ Qp, const __bf16* __restrict__ Kp,
                       const __bf16* __restrict__ Vp, const int* __restrict__ mask,
                       __bf16* __restrict__ Ctx) {
    constexpr float SCALE = 0.08838834764831845f;   // 1/sqrt(128)

    __shared__ __bf16 Ks[64][136];     // keys x HD (+8 pad)
    __shared__ __bf16 Vt[128][72];     // HD x keys (+8 pad), transposed V
    __shared__ __bf16 Ps[8][16][72];   // per-wave P tile: 16 rows x 64 keys (+pad)

    const int qb = blockIdx.x, h = blockIdx.y, b = blockIdx.z;
    const int t = threadIdx.x;
    const int w = t >> 5, lane = t & 31;
    const int ll = lane & 15, lh = lane >> 4;
    const int qr0 = qb * 128 + w * 16;

    // resident Q fragments: 16 rows x 128 (4 WMMA K-steps of 32)
    AFrag qf[4];
    {
        const __bf16* qrow = Qp + (size_t)(b * kSQ + qr0 + ll) * kE + h * kHD;
        for (int s = 0; s < 4; ++s) {
            qf[s].h[0] = *(const v8bf*)(qrow + s * 32 + lh * 8);
            qf[s].h[1] = *(const v8bf*)(qrow + s * 32 + 16 + lh * 8);
        }
    }

    v8f oacc[8];
    float m[8], lsum[8];
    for (int c = 0; c < 8; ++c) oacc[c] = vzero8();
    for (int r = 0; r < 8; ++r) { m[r] = -3.0e38f; lsum[r] = 0.f; }

    for (int kb = 0; kb < kSK / 64; ++kb) {
        __syncthreads();   // prior tile's LDS reads done before overwriting
        {
            const int r  = t >> 2;
            const int cc = (t & 3) * 32;
            const __bf16* ksrc = Kp + (size_t)(b * kSK + kb * 64 + r) * kE + h * kHD + cc;
            async_copy16(&Ks[r][cc],      ksrc);          // ASYNCcnt path
            async_copy16(&Ks[r][cc + 8],  ksrc + 8);
            async_copy16(&Ks[r][cc + 16], ksrc + 16);
            async_copy16(&Ks[r][cc + 24], ksrc + 24);
            const __bf16* vsrc = Vp + (size_t)(b * kSK + kb * 64 + r) * kE + h * kHD + cc;
            for (int i = 0; i < 32; ++i) Vt[cc + i][r] = vsrc[i];   // transpose staging
        }
        wait_async_le<0>();
        __syncthreads();

        // S = Q * K^T  (4 column subtiles of 16 keys)
        v8f St[4];
        for (int j = 0; j < 4; ++j) {
            v8f s = vzero8();
            for (int ks = 0; ks < 4; ++ks) {
                AFrag kf;
                kf.h[0] = *(const v8bf*)(&Ks[j * 16 + ll][ks * 32 + lh * 8]);
                kf.h[1] = *(const v8bf*)(&Ks[j * 16 + ll][ks * 32 + 16 + lh * 8]);
                s = __builtin_amdgcn_wmma_f32_16x16x32_bf16(
                    false, qf[ks].v, false, kf.v, (short)0, s, false, false);
            }
            const int kcol = kb * 64 + j * 16 + ll;
            for (int r = 0; r < 8; ++r) {
                const int mk = mask[(size_t)(b * kSQ + qr0 + r + 8 * lh) * kSK + kcol];
                St[j][r] = (mk == 0) ? -1.0e10f : s[r] * SCALE;
            }
        }

        // online softmax: row stats share the C-tile lane layout (vgpr r <-> row r/r+8)
        float alpha[8];
        for (int r = 0; r < 8; ++r) {
            float tm = fmaxf(fmaxf(St[0][r], St[1][r]), fmaxf(St[2][r], St[3][r]));
            for (int d = 1; d < 16; d <<= 1) tm = fmaxf(tm, __shfl_xor(tm, d, 32));
            const float nm = fmaxf(m[r], tm);
            alpha[r] = __expf(m[r] - nm);
            m[r] = nm;
            lsum[r] *= alpha[r];
        }
        for (int c = 0; c < 8; ++c)
            for (int r = 0; r < 8; ++r) oacc[c][r] *= alpha[r];

        for (int r = 0; r < 8; ++r) {
            float ps = 0.f;
            for (int j = 0; j < 4; ++j) {
                const float p = __expf(St[j][r] - m[r]);
                ps += p;
                Ps[w][r + 8 * lh][j * 16 + ll] = (__bf16)p;
            }
            for (int d = 1; d < 16; d <<= 1) ps += __shfl_xor(ps, d, 32);
            lsum[r] += ps;
        }

        // O += P * V  (A = P from per-wave LDS, B = V^T fragments from Vt)
        for (int kk = 0; kk < 2; ++kk) {
            AFrag pf;
            pf.h[0] = *(const v8bf*)(&Ps[w][ll][kk * 32 + lh * 8]);
            pf.h[1] = *(const v8bf*)(&Ps[w][ll][kk * 32 + 16 + lh * 8]);
            for (int c = 0; c < 8; ++c) {
                AFrag vf;
                vf.h[0] = *(const v8bf*)(&Vt[c * 16 + ll][kk * 32 + lh * 8]);
                vf.h[1] = *(const v8bf*)(&Vt[c * 16 + ll][kk * 32 + 16 + lh * 8]);
                oacc[c] = __builtin_amdgcn_wmma_f32_16x16x32_bf16(
                    false, pf.v, false, vf.v, (short)0, oacc[c], false, false);
            }
        }
    }

    for (int r = 0; r < 8; ++r) {
        const float inv = 1.f / lsum[r];
        const size_t row = (size_t)(b * kSQ + qr0 + r + 8 * lh);
        for (int c = 0; c < 8; ++c)
            Ctx[row * kE + h * kHD + c * 16 + ll] = (__bf16)(oacc[c][r] * inv);
    }
}

// ---------------- host launch ----------------
extern "C" void kernel_launch(void* const* d_in, const int* in_sizes, int n_in,
                              void* d_out, int out_size, void* d_ws, size_t ws_size,
                              hipStream_t stream) {
    (void)in_sizes; (void)n_in; (void)out_size; (void)ws_size;

    const float* query = (const float*)d_in[0];
    const float* key   = (const float*)d_in[1];
    const float* value = (const float*)d_in[2];
    const int*   mask  = (const int*)d_in[3];
    const float* Wq = (const float*)d_in[4];
    const float* bq = (const float*)d_in[5];
    const float* Wk = (const float*)d_in[6];
    const float* bk = (const float*)d_in[7];
    const float* Wv = (const float*)d_in[8];
    const float* bv = (const float*)d_in[9];
    const float* Wo = (const float*)d_in[10];
    const float* bo = (const float*)d_in[11];

    const size_t M   = (size_t)kB * kSQ;            // 8192
    const size_t NQ  = M * kQD;
    const size_t NKV = (size_t)kB * kSK * kKVD;
    const size_t NWQ = (size_t)kE * kQD;
    const size_t NWK = (size_t)kE * kKVD;
    const size_t NE  = M * kE;

    char* ws = (char*)d_ws;
    size_t off = 0;
    auto take = [&](size_t elems) { __bf16* p = (__bf16*)(ws + off); off += elems * 2; return p; };

    __bf16* q_bf  = take(NQ);
    __bf16* k_bf  = take(NKV);
    __bf16* v_bf  = take(NKV);
    __bf16* wq_bf = take(NWQ);
    __bf16* wk_bf = take(NWK);
    __bf16* wv_bf = take(NWK);
    __bf16* wo_bf = take(NWQ);
    __bf16* Qp    = take(NE);
    __bf16* Kp    = take(NE);
    __bf16* Vp    = take(NE);
    __bf16* Ctx   = take(NE);

    auto cvt = [&](const float* s, __bf16* d, size_t n) {
        const int n4 = (int)(n / 4);
        cvt_f32_bf16_kernel<<<dim3((n4 + 255) / 256), dim3(256), 0, stream>>>(s, d, n4);
    };
    cvt(query, q_bf, NQ);
    cvt(key,   k_bf, NKV);
    cvt(value, v_bf, NKV);
    cvt(Wq, wq_bf, NWQ);
    cvt(Wk, wk_bf, NWK);
    cvt(Wv, wv_bf, NWK);
    cvt(Wo, wo_bf, NWQ);

    // projections: C[M, E] = X[M, K] * W[E, K]^T + b
    gemm_nt_kernel<true><<<dim3(kE / 128, (int)(M / 256)), dim3(256), 0, stream>>>(
        q_bf, wq_bf, bq, (void*)Qp, (int)M, kE, kQD);
    gemm_nt_kernel<true><<<dim3(kE / 128, (int)(M / 256)), dim3(256), 0, stream>>>(
        k_bf, wk_bf, bk, (void*)Kp, (int)M, kE, kKVD);
    gemm_nt_kernel<true><<<dim3(kE / 128, (int)(M / 256)), dim3(256), 0, stream>>>(
        v_bf, wv_bf, bv, (void*)Vp, (int)M, kE, kKVD);

    flash_attn_kernel<<<dim3(kSQ / 128, kH, kB), dim3(256), 0, stream>>>(
        Qp, Kp, Vp, mask, Ctx);

    // output projection to f32 d_out
    gemm_nt_kernel<false><<<dim3(kE / 128, (int)(M / 256)), dim3(256), 0, stream>>>(
        Ctx, wo_bf, bo, d_out, (int)M, kE, kE);
}
